// MultiScaleGatedRecurrence_76459007803795
// MI455X (gfx1250) — matrix operand order repaired and checked by reading, not compiled, not run on hardware
//
#include <hip/hip_runtime.h>
#include <hip/hip_bf16.h>
#include <math.h>

#define D_MODEL 1024
#define D_REC   1024
#define BATCH   4
#define SEQ     8192
#define ROWS    (BATCH * SEQ)        // 32768
#define NCHUNK  32
#define CHLEN   (SEQ / NCHUNK)       // 256

typedef __attribute__((ext_vector_type(16))) __bf16    v16bf;
typedef __attribute__((ext_vector_type(8)))  float     v8f;
typedef __attribute__((ext_vector_type(8)))  unsigned  v8u;

#define KT    32          // K-step per LDS tile == WMMA K
#define BM    64
#define BN    64
#define LDSX  36          // fp32 stride for async-staged x tile (16B-aligned segs)
#define LDK2  17          // u32 (bf16-pair) stride for W tiles (odd -> bank spread)

#define LDS_OFF(p) ((unsigned)(unsigned long long)(p))

// pack two fp32 -> one u32 holding {bf16(lo), bf16(hi)}
__device__ __forceinline__ unsigned pk2(float lo, float hi) {
  union { __bf16 h[2]; unsigned u; } v;
  v.h[0] = (__bf16)lo; v.h[1] = (__bf16)hi;
  return v.u;
}

// ---- async global -> LDS copy of a BM x KT fp32 tile (GV mode, b128) -----
// 64 rows * 128B = 512 16B-chunks, 128 threads * 4 issues, tracked by ASYNCcnt
__device__ __forceinline__ void issue_x_async(const float* gbase, int gstride,
                                              unsigned ldsBase, int tid) {
#pragma unroll
  for (int i = 0; i < 4; ++i) {
    int q   = tid + i * 128;
    int row = q >> 3;
    int seg = q & 7;
    const float* g = gbase + (size_t)row * gstride + seg * 4;
    unsigned l = ldsBase + (unsigned)((row * LDSX + seg * 4) * sizeof(float));
    asm volatile("global_load_async_to_lds_b128 %0, %1, off"
                 :: "v"(l), "v"(g) : "memory");
  }
}
__device__ __forceinline__ void wait_async0() {
  asm volatile("s_wait_asynccnt 0x0" ::: "memory");
}

// ---- WMMA fragments (wave32), this code's contiguous k order for both A and B ------------------------
// A (16x32): M = lane&15 ; elem e -> K=(lane>>4)*16+e
// source is fp32 in LDS, convert while building the fragment
__device__ __forceinline__ v16bf ld_a_frag_f32(const float* s, int lane) {
  int l = lane & 15, hi = lane >> 4;
  const float* r = s + l * LDSX;
  v16bf f;
#pragma unroll
  for (int e = 0; e < 8; ++e) f[e] = (__bf16)r[hi * 16 + e];
#pragma unroll
  for (int e = 0; e < 8; ++e) f[8 + e] = (__bf16)r[hi * 16 + 8 + e];
  return f;
}
// B (32x16): N = lane&15 ; K = (lane>>4)*16 + e ; stored as bf16-pairs [n][kpair]
__device__ __forceinline__ v16bf ld_b_frag_pk(const unsigned* s, int lane) {
  int l = lane & 15, hi = lane >> 4;
  const unsigned* c = s + l * LDK2 + hi * 8;
  union { v8u u; v16bf f; } w;
#pragma unroll
  for (int j = 0; j < 8; ++j) w.u[j] = c[j];
  return w.f;
}

__device__ __forceinline__ float sigmoidf_(float x) {
  return 1.0f / (1.0f + __expf(-x));
}

// ---- Kernel 1: aiv = x @ W_aiv, fused gating -> A (decay) and U (input) --
__global__ __launch_bounds__(128)
void k_gemm_gate(const float* __restrict__ x, const float* __restrict__ Waiv,
                 const float* __restrict__ bias,
                 float* __restrict__ Abuf, float* __restrict__ Ubuf) {
  __shared__ float    sXf[2][BM * LDSX];   // async-staged x tiles (double buffer)
  __shared__ unsigned sWp[3][BN * LDK2];   // W tiles (a,i,v) as bf16 pairs, [n][kpair]

  const int row0 = blockIdx.x * BM;
  const int col0 = blockIdx.y * BN;
  const int tid  = threadIdx.x;
  const int wave = tid >> 5;
  const int lane = tid & 31;

  v8f accA[4], accI[4], accV[4];
#pragma unroll
  for (int nb = 0; nb < 4; ++nb) { accA[nb] = {}; accI[nb] = {}; accV[nb] = {}; }

  const float* xrow = x + (size_t)row0 * D_MODEL;
  issue_x_async(xrow, D_MODEL, LDS_OFF(&sXf[0][0]), tid);

  for (int k0 = 0; k0 < D_MODEL; k0 += KT) {
    const int cur = (k0 >> 5) & 1;

    // stage W tiles: 3 x (32 x 64) fp32 -> packed bf16 pairs, transposed [n][kpair]
    {
      int n  = tid & 63;
      int kb = tid >> 6;  // 0..1
#pragma unroll
      for (int g = 0; g < 3; ++g) {
        const float* gw = Waiv + (size_t)k0 * (3 * D_REC) + g * D_REC + col0 + n;
        if (k0 + KT < D_MODEL) __builtin_prefetch(gw + (size_t)KT * (3 * D_REC), 0, 0);
#pragma unroll
        for (int jp = kb; jp < KT / 2; jp += 2) {
          float lo = gw[(size_t)(2 * jp)     * (3 * D_REC)];
          float hi = gw[(size_t)(2 * jp + 1) * (3 * D_REC)];
          sWp[g][n * LDK2 + jp] = pk2(lo, hi);
        }
      }
    }

    wait_async0();        // this tile's x is in LDS (own wave)
    __syncthreads();      // ...and everyone else's chunks too

    if (k0 + KT < D_MODEL)  // stream next x tile into the other buffer
      issue_x_async(xrow + (k0 + KT), D_MODEL, LDS_OFF(&sXf[cur ^ 1][0]), tid);

    v16bf af = ld_a_frag_f32(&sXf[cur][(wave * 16) * LDSX], lane);
#pragma unroll
    for (int nb = 0; nb < 4; ++nb) {
      v16bf bA = ld_b_frag_pk(&sWp[0][(nb * 16) * LDK2], lane);
      v16bf bI = ld_b_frag_pk(&sWp[1][(nb * 16) * LDK2], lane);
      v16bf bV = ld_b_frag_pk(&sWp[2][(nb * 16) * LDK2], lane);
      accA[nb] = __builtin_amdgcn_wmma_f32_16x16x32_bf16(false, af, false, bA,
                   (short)0, accA[nb], false, false);
      accI[nb] = __builtin_amdgcn_wmma_f32_16x16x32_bf16(false, af, false, bI,
                   (short)0, accI[nb], false, false);
      accV[nb] = __builtin_amdgcn_wmma_f32_16x16x32_bf16(false, af, false, bV,
                   (short)0, accV[nb], false, false);
    }
    __syncthreads();      // protect sWp before next iteration overwrites it
  }

  // epilogue: gate and store decay a + gated input u
  const int l = lane & 15, hi = lane >> 4;
#pragma unroll
  for (int nb = 0; nb < 4; ++nb) {
    int col = col0 + nb * 16 + l;
    float db = bias[col];
#pragma unroll
    for (int r = 0; r < 8; ++r) {
      int row = row0 + wave * 16 + hi * 8 + r;
      float a  = sigmoidf_(accA[nb][r] + db);
      float ig = sigmoidf_(accI[nb][r]);
      float u  = sqrtf(fmaxf(1.0f - a * a, 1e-8f)) * (ig * accV[nb][r]);
      size_t idx = (size_t)row * D_REC + col;
      Abuf[idx] = a;
      Ubuf[idx] = u;
    }
  }
}

// ---- Kernel 2: per-(channel,chunk) decay product + zero-init local state -
__global__ __launch_bounds__(256)
void k_scan_partial(const float* __restrict__ A, const float* __restrict__ U,
                    float* __restrict__ P, float* __restrict__ S) {
  int t = blockIdx.x * blockDim.x + threadIdx.x;   // 0 .. BATCH*NCHUNK*D_REC-1
  int d = t & (D_REC - 1);
  int c = (t >> 10) & (NCHUNK - 1);
  int b = t >> 15;
  size_t base = ((size_t)b * SEQ + (size_t)c * CHLEN) * D_REC + d;
  float p = 1.0f, s = 0.0f;
  for (int i = 0; i < CHLEN; ++i) {
    float aa = A[base + (size_t)i * D_REC];
    float uu = U[base + (size_t)i * D_REC];
    s = fmaf(aa, s, uu);
    p *= aa;
  }
  int ch = b * D_REC + d;
  P[(size_t)ch * NCHUNK + c] = p;
  S[(size_t)ch * NCHUNK + c] = s;
}

// ---- Kernel 3: sequential inter-chunk combine (tiny) ---------------------
__global__ __launch_bounds__(256)
void k_scan_combine(const float* __restrict__ P, const float* __restrict__ S,
                    float* __restrict__ Cin) {
  int ch = blockIdx.x * blockDim.x + threadIdx.x;  // 0 .. BATCH*D_REC-1
  const float* p = P + (size_t)ch * NCHUNK;
  const float* s = S + (size_t)ch * NCHUNK;
  float* ci = Cin + (size_t)ch * NCHUNK;
  float carry = 0.0f;
  for (int c = 0; c < NCHUNK; ++c) {
    ci[c] = carry;
    carry = fmaf(p[c], carry, s[c]);
  }
}

// ---- Kernel 4: rerun recurrence seeded with incoming state, U <- h -------
__global__ __launch_bounds__(256)
void k_scan_apply(const float* __restrict__ A, float* __restrict__ U,
                  const float* __restrict__ Cin) {
  int t = blockIdx.x * blockDim.x + threadIdx.x;
  int d = t & (D_REC - 1);
  int c = (t >> 10) & (NCHUNK - 1);
  int b = t >> 15;
  int ch = b * D_REC + d;
  size_t base = ((size_t)b * SEQ + (size_t)c * CHLEN) * D_REC + d;
  float h = Cin[(size_t)ch * NCHUNK + c];
  for (int i = 0; i < CHLEN; ++i) {
    size_t idx = base + (size_t)i * D_REC;
    h = fmaf(A[idx], h, U[idx]);
    U[idx] = h;   // overwrite u with h in place
  }
}

// ---- Kernel 5: out = h @ W_mix + b_mix -----------------------------------
__global__ __launch_bounds__(128)
void k_gemm_mix(const float* __restrict__ h, const float* __restrict__ Wmix,
                const float* __restrict__ bmix, float* __restrict__ out) {
  __shared__ float    sXf[2][BM * LDSX];
  __shared__ unsigned sWp[BN * LDK2];

  const int row0 = blockIdx.x * BM;
  const int col0 = blockIdx.y * BN;
  const int tid  = threadIdx.x;
  const int wave = tid >> 5;
  const int lane = tid & 31;

  v8f acc[4];
#pragma unroll
  for (int nb = 0; nb < 4; ++nb) acc[nb] = {};

  const float* hrow = h + (size_t)row0 * D_REC;
  issue_x_async(hrow, D_REC, LDS_OFF(&sXf[0][0]), tid);

  for (int k0 = 0; k0 < D_REC; k0 += KT) {
    const int cur = (k0 >> 5) & 1;
    {
      int n  = tid & 63;
      int kb = tid >> 6;
      const float* gw = Wmix + (size_t)k0 * D_MODEL + col0 + n;
      if (k0 + KT < D_REC) __builtin_prefetch(gw + (size_t)KT * D_MODEL, 0, 0);
#pragma unroll
      for (int jp = kb; jp < KT / 2; jp += 2) {
        float lo = gw[(size_t)(2 * jp)     * D_MODEL];
        float hi = gw[(size_t)(2 * jp + 1) * D_MODEL];
        sWp[n * LDK2 + jp] = pk2(lo, hi);
      }
    }

    wait_async0();
    __syncthreads();

    if (k0 + KT < D_REC)
      issue_x_async(hrow + (k0 + KT), D_REC, LDS_OFF(&sXf[cur ^ 1][0]), tid);

    v16bf af = ld_a_frag_f32(&sXf[cur][(wave * 16) * LDSX], lane);
#pragma unroll
    for (int nb = 0; nb < 4; ++nb) {
      v16bf bf = ld_b_frag_pk(&sWp[(nb * 16) * LDK2], lane);
      acc[nb] = __builtin_amdgcn_wmma_f32_16x16x32_bf16(false, af, false, bf,
                  (short)0, acc[nb], false, false);
    }
    __syncthreads();
  }

  const int l = lane & 15, hi = lane >> 4;
#pragma unroll
  for (int nb = 0; nb < 4; ++nb) {
    int col = col0 + nb * 16 + l;
    float bb = bmix[col];
#pragma unroll
    for (int r = 0; r < 8; ++r) {
      int row = row0 + wave * 16 + hi * 8 + r;
      out[(size_t)row * D_MODEL + col] = acc[nb][r] + bb;
    }
  }
}

extern "C" void kernel_launch(void* const* d_in, const int* in_sizes, int n_in,
                              void* d_out, int out_size, void* d_ws, size_t ws_size,
                              hipStream_t stream) {
  (void)in_sizes; (void)n_in; (void)out_size; (void)ws_size;
  const float* x    = (const float*)d_in[0];
  const float* Waiv = (const float*)d_in[1];
  const float* bias = (const float*)d_in[2];
  const float* Wmix = (const float*)d_in[3];
  const float* bmix = (const float*)d_in[4];
  float* out = (float*)d_out;

  float* ws = (float*)d_ws;
  const size_t N  = (size_t)ROWS * D_REC;              // 33.5M floats per stream
  const size_t NC = (size_t)BATCH * D_REC * NCHUNK;    // 131072
  float* Abuf = ws;            // decay a      (128 MB)
  float* Ubuf = ws + N;        // u, then h    (128 MB)
  float* P    = Ubuf + N;
  float* S    = P + NC;
  float* Cin  = S + NC;

  dim3 g1(ROWS / BM, D_REC / BN);    // 512 x 16
  k_gemm_gate<<<g1, 128, 0, stream>>>(x, Waiv, bias, Abuf, Ubuf);

  int scanThreads = BATCH * NCHUNK * D_REC;            // 131072
  k_scan_partial<<<scanThreads / 256, 256, 0, stream>>>(Abuf, Ubuf, P, S);
  k_scan_combine<<<(BATCH * D_REC) / 256, 256, 0, stream>>>(P, S, Cin);
  k_scan_apply<<<scanThreads / 256, 256, 0, stream>>>(Abuf, Ubuf, Cin);

  dim3 g2(ROWS / BM, D_MODEL / BN);  // 512 x 16
  k_gemm_mix<<<g2, 128, 0, stream>>>(Ubuf, Wmix, bmix, out);
}
